// CCAR_11579231830663
// MI455X (gfx1250) — compile-verified, matrix operand order checked by reading, not run on hardware
//
#include <hip/hip_runtime.h>
#include <math.h>

// Problem constants (match reference)
#define BB 16
#define CC 512
#define WW 1024
static const long long NBCW = (long long)BB * CC * WW;   // 8,388,608
static const long long NWW2 = (long long)BB * WW * WW;   // 16,777,216

typedef __attribute__((ext_vector_type(16))) __bf16 v16bf;
typedef __attribute__((ext_vector_type(8)))  __bf16 v8bf;
typedef __attribute__((ext_vector_type(8)))  float  v8f;

// Load a 16-bit WMMA A/B fragment when K is the contiguous (innermost) axis.
// Per ISA 7.12.2 the lane's 16 elements are two contiguous 8-element chunks:
// k in [8*half, 8*half+8) and [16+8*half, 16+8*half+8)  -> two b128 loads.
__device__ __forceinline__ v16bf load_frag_kcontig(const __bf16* __restrict__ p, int half) {
    v8bf c0 = *(const v8bf*)(p + 8 * half);
    v8bf c1 = *(const v8bf*)(p + 16 + 8 * half);
    v16bf f;
#pragma unroll
    for (int i = 0; i < 8; ++i) { f[i] = c0[i]; f[i + 8] = c1[i]; }
    return f;
}

// ---------------- elementwise / pack kernels ----------------

__global__ void cvt_bf16_kernel(const float* __restrict__ x, __bf16* __restrict__ y, long long n) {
    long long i = (long long)blockIdx.x * blockDim.x + threadIdx.x;
    if (i < n) y[i] = (__bf16)x[i];
}

// rw [C, C, 3] (co, ci, k) -> wp [3, C, C] (k, co, ci), bf16 (K=ci contiguous)
__global__ void pack_conv3_kernel(const float* __restrict__ w, __bf16* __restrict__ wp) {
    long long i = (long long)blockIdx.x * blockDim.x + threadIdx.x;
    long long n = 3LL * CC * CC;
    if (i >= n) return;
    int k  = (int)(i % 3);
    int ci = (int)((i / 3) % CC);
    int co = (int)(i / (3LL * CC));
    wp[(long long)k * CC * CC + (long long)co * CC + ci] = (__bf16)w[i];
}

__global__ void add_xg_kernel(const float* __restrict__ x, const float* __restrict__ g,
                              float* __restrict__ xg, long long n) {
    long long i = (long long)blockIdx.x * blockDim.x + threadIdx.x;
    if (i < n) xg[i] = x[i] + g[i];
}

__global__ void zero_sums_kernel(double* sums) {
    if (threadIdx.x < 4) sums[threadIdx.x] = 0.0;
}

__global__ void reduce_sums_kernel(const float* __restrict__ x, long long n, double* __restrict__ sums) {
    __shared__ double s0[256];
    __shared__ double s1[256];
    int tid = threadIdx.x;
    double a = 0.0, b = 0.0;
    for (long long i = (long long)blockIdx.x * blockDim.x + tid; i < n;
         i += (long long)gridDim.x * blockDim.x) {
        double v = (double)x[i];
        a += v;
        b += v * v;
    }
    s0[tid] = a; s1[tid] = b;
    __syncthreads();
    for (int s = 128; s > 0; s >>= 1) {
        if (tid < s) { s0[tid] += s0[tid + s]; s1[tid] += s1[tid + s]; }
        __syncthreads();
    }
    if (tid == 0) {
        atomicAdd(&sums[0], s0[0]);
        atomicAdd(&sums[1], s1[0]);
    }
}

// LDS-tiled transpose + f32->bf16 convert: in [B][C][W] f32 -> out [B][W][C] bf16.
// If sums!=null also applies global z-score (mean over all, std ddof=1).
__global__ void transpose_cvt_kernel(const float* __restrict__ in, __bf16* __restrict__ out,
                                     const double* __restrict__ sums) {
    __shared__ float tile[32][33];
    const long long CW = (long long)CC * WW;
    int b  = blockIdx.z;
    int w0 = blockIdx.x * 32;   // along W
    int c0 = blockIdx.y * 32;   // along C
    const float* ib = in + (long long)b * CW;
#pragma unroll
    for (int r = 0; r < 32; r += 8)
        tile[threadIdx.y + r][threadIdx.x] =
            ib[(long long)(c0 + threadIdx.y + r) * WW + w0 + threadIdx.x];
    __syncthreads();
    float mu = 0.f, inv = 1.f;
    if (sums) {
        double s0 = sums[0], s1 = sums[1];
        double n  = (double)NBCW;
        double mean = s0 / n;
        double var  = (s1 - s0 * s0 / n) / (n - 1.0);
        mu  = (float)mean;
        inv = (float)(1.0 / sqrt(var));
    }
    __bf16* ob = out + (long long)b * CW;
#pragma unroll
    for (int r = 0; r < 32; r += 8)
        ob[(long long)(w0 + threadIdx.y + r) * CC + c0 + threadIdx.x] =
            (__bf16)((tile[threadIdx.x][threadIdx.y + r] - mu) * inv);
}

// per-(b,c) instance norm over W (biased var, eps=1e-5) then sin (in-place capable)
__global__ void in_sin_kernel(const float* __restrict__ y, float* __restrict__ outf) {
    __shared__ float s0[256];
    __shared__ float s1[256];
    int bc  = blockIdx.x;                 // 0 .. B*C-1
    int tid = threadIdx.x;
    const float* row = y + (long long)bc * WW;
    float v[4];
    float a = 0.f, b2 = 0.f;
#pragma unroll
    for (int s = 0; s < 4; ++s) {
        v[s] = row[tid + 256 * s];
        a += v[s]; b2 += v[s] * v[s];
    }
    s0[tid] = a; s1[tid] = b2;
    __syncthreads();
    for (int s = 128; s > 0; s >>= 1) {
        if (tid < s) { s0[tid] += s0[tid + s]; s1[tid] += s1[tid + s]; }
        __syncthreads();
    }
    float mean = s0[0] * (1.0f / WW);
    float var  = s1[0] * (1.0f / WW) - mean * mean;
    float rstd = rsqrtf(var + 1e-5f);
#pragma unroll
    for (int s = 0; s < 4; ++s)
        outf[(long long)bc * WW + tid + 256 * s] = sinf((v[s] - mean) * rstd);
}

// row softmax over last dim of [B*W, W]; writes bf16 attention rows (K-contiguous)
__global__ void softmax_att_kernel(const float* __restrict__ energy, __bf16* __restrict__ attb) {
    __shared__ float sh[256];
    long long row = blockIdx.x;           // 0 .. B*W-1
    int tid = threadIdx.x;
    const float* e = energy + row * (long long)WW;
    float v[4];
    float m = -INFINITY;
#pragma unroll
    for (int s = 0; s < 4; ++s) { v[s] = e[tid + 256 * s]; m = fmaxf(m, v[s]); }
    sh[tid] = m;
    __syncthreads();
    for (int s = 128; s > 0; s >>= 1) {
        if (tid < s) sh[tid] = fmaxf(sh[tid], sh[tid + s]);
        __syncthreads();
    }
    m = sh[0];
    __syncthreads();
    float sum = 0.f;
#pragma unroll
    for (int s = 0; s < 4; ++s) { v[s] = __expf(v[s] - m); sum += v[s]; }
    sh[tid] = sum;
    __syncthreads();
    for (int s = 128; s > 0; s >>= 1) {
        if (tid < s) sh[tid] += sh[tid + s];
        __syncthreads();
    }
    float inv = 1.0f / sh[0];
#pragma unroll
    for (int s = 0; s < 4; ++s)
        attb[row * (long long)WW + tid + 256 * s] = (__bf16)(v[s] * inv);
}

// ---------------- WMMA kernels ----------------

// Batched GEMM with K-contiguous operands:
//   D[m,n] = sum_k A[m*a_m + k] * Bt[n*b_n + k]
// One wave computes a 16x64 tile (4 accumulators, A fragment reused x4).
__global__ void gemm2_wmma_bf16_kernel(
    const __bf16* __restrict__ A, long long a_bat, long long a_m,
    const __bf16* __restrict__ Bt, long long b_bat, long long b_n,
    float* __restrict__ Cf, __bf16* __restrict__ Cb,
    long long c_bat, long long c_m, long long c_n, int K) {
    int lane = threadIdx.x;
    int half = lane >> 4;
    int lo   = lane & 15;
    int n0   = blockIdx.x * 64;
    int m0   = blockIdx.y * 16;
    long long bat = blockIdx.z;
    const __bf16* Ap = A + bat * a_bat + (long long)(m0 + lo) * a_m;
    const __bf16* Bp0 = Bt + bat * b_bat + (long long)(n0 + lo) * b_n;
    v8f acc0 = {}, acc1 = {}, acc2 = {}, acc3 = {};
    for (int k0 = 0; k0 < K; k0 += 32) {
        v16bf af = load_frag_kcontig(Ap + k0, half);
        v16bf b0 = load_frag_kcontig(Bp0 + k0, half);
        v16bf b1 = load_frag_kcontig(Bp0 + 16 * b_n + k0, half);
        v16bf b2 = load_frag_kcontig(Bp0 + 32 * b_n + k0, half);
        v16bf b3 = load_frag_kcontig(Bp0 + 48 * b_n + k0, half);
        acc0 = __builtin_amdgcn_wmma_f32_16x16x32_bf16(false, af, false, b0, (short)0, acc0, false, false);
        acc1 = __builtin_amdgcn_wmma_f32_16x16x32_bf16(false, af, false, b1, (short)0, acc1, false, false);
        acc2 = __builtin_amdgcn_wmma_f32_16x16x32_bf16(false, af, false, b2, (short)0, acc2, false, false);
        acc3 = __builtin_amdgcn_wmma_f32_16x16x32_bf16(false, af, false, b3, (short)0, acc3, false, false);
    }
    v8f accs[4] = {acc0, acc1, acc2, acc3};
#pragma unroll
    for (int s = 0; s < 4; ++s) {
#pragma unroll
        for (int r = 0; r < 8; ++r) {
            long long idx = bat * c_bat + (long long)(m0 + r + 8 * half) * c_m
                          + (long long)(n0 + 16 * s + lo) * c_n;
            if (Cf) Cf[idx] = accs[s][r];
            if (Cb) Cb[idx] = (__bf16)accs[s][r];
        }
    }
}

// 3-tap conv over W via WMMA: out[b,co,w] = sum_t sum_ci Wt[co,ci]*x[b,ci,w+t-1]
// A = wp[t] rows (ci contiguous); Bt = xT [B][W][C] rows (ci contiguous).
__global__ void conv3_wmma_kernel(const __bf16* __restrict__ xT, const __bf16* __restrict__ wp,
                                  float* __restrict__ out) {
    int lane = threadIdx.x;
    int half = lane >> 4;
    int lo   = lane & 15;
    int w0 = blockIdx.x * 64;
    int c0 = blockIdx.y * 16;
    int b  = blockIdx.z;
    const long long CW = (long long)CC * WW;
    const __bf16* xb = xT + (long long)b * CW;
    v8f acc0 = {}, acc1 = {}, acc2 = {}, acc3 = {};
    v16bf zf = {};
    for (int t = 0; t < 3; ++t) {
        const __bf16* Ap = wp + (long long)t * CC * CC + (long long)(c0 + lo) * CC;
        int wi0 = w0 + lo + t - 1;
        bool ok0 = (wi0 >= 0) && (wi0 < WW);
        bool ok1 = (wi0 + 16 >= 0) && (wi0 + 16 < WW);
        bool ok2 = (wi0 + 32 >= 0) && (wi0 + 32 < WW);
        bool ok3 = (wi0 + 48 >= 0) && (wi0 + 48 < WW);
        const __bf16* r0 = xb + (long long)wi0 * CC;
        for (int k0 = 0; k0 < CC; k0 += 32) {
            v16bf af = load_frag_kcontig(Ap + k0, half);
            v16bf b0 = ok0 ? load_frag_kcontig(r0 + k0, half) : zf;
            v16bf b1 = ok1 ? load_frag_kcontig(r0 + 16LL * CC + k0, half) : zf;
            v16bf b2 = ok2 ? load_frag_kcontig(r0 + 32LL * CC + k0, half) : zf;
            v16bf b3 = ok3 ? load_frag_kcontig(r0 + 48LL * CC + k0, half) : zf;
            acc0 = __builtin_amdgcn_wmma_f32_16x16x32_bf16(false, af, false, b0, (short)0, acc0, false, false);
            acc1 = __builtin_amdgcn_wmma_f32_16x16x32_bf16(false, af, false, b1, (short)0, acc1, false, false);
            acc2 = __builtin_amdgcn_wmma_f32_16x16x32_bf16(false, af, false, b2, (short)0, acc2, false, false);
            acc3 = __builtin_amdgcn_wmma_f32_16x16x32_bf16(false, af, false, b3, (short)0, acc3, false, false);
        }
    }
    v8f accs[4] = {acc0, acc1, acc2, acc3};
#pragma unroll
    for (int s = 0; s < 4; ++s) {
#pragma unroll
        for (int r = 0; r < 8; ++r) {
            int m = c0 + r + 8 * half;
            out[(long long)b * CW + (long long)m * WW + w0 + 16 * s + lo] = accs[s][r];
        }
    }
}

// ---------------- driver ----------------

extern "C" void kernel_launch(void* const* d_in, const int* in_sizes, int n_in,
                              void* d_out, int out_size, void* d_ws, size_t ws_size,
                              hipStream_t stream) {
    (void)in_sizes; (void)n_in; (void)out_size; (void)ws_size;
    const float* x   = (const float*)d_in[0];
    const float* qw  = (const float*)d_in[1];
    const float* kw  = (const float*)d_in[3];
    const float* vw  = (const float*)d_in[5];
    const float* rw1 = (const float*)d_in[7];
    const float* rw2 = (const float*)d_in[9];
    float* out = (float*)d_out;

    const long long CW = (long long)CC * WW;
    const long long W2 = (long long)WW * WW;

    // workspace layout
    char* p = (char*)d_ws;
    __bf16* xT    = (__bf16*)p;              p += NBCW * 2;   // [B][W][C]
    __bf16* wr1p  = (__bf16*)p;              p += 3LL * CC * CC * 2;
    __bf16* wr2p  = (__bf16*)p;              p += 3LL * CC * CC * 2;
    __bf16* wq    = (__bf16*)p;              p += (long long)CC * CC * 2;
    __bf16* wk    = (__bf16*)p;              p += (long long)CC * CC * 2;
    __bf16* wv    = (__bf16*)p;              p += (long long)CC * CC * 2;
    float*  t0    = (float*)p;               p += NBCW * 4;   // conv scratch; energy alias
    float*  xg    = (float*)p;               p += NBCW * 4;   // adjacent to t0 (energy alias)
    float*  g     = (float*)p;               p += NBCW * 4;
    __bf16* g1T   = (__bf16*)p;              p += NBCW * 2;   // [B][W][C]
    __bf16* gT    = (__bf16*)p;              p += NBCW * 2;   // [B][W][C]
    __bf16* xgT   = (__bf16*)p;              p += NBCW * 2;   // [B][W][C]
    __bf16* xzT   = (__bf16*)p;              p += NBCW * 2;   // [B][W][C]
    __bf16* gzT   = (__bf16*)p;              p += NBCW * 2;   // [B][W][C]
    __bf16* pqT   = (__bf16*)p;              p += NBCW * 2;   // [B][W][C] (pq[c,j] -> [j][c])
    __bf16* pkT   = (__bf16*)p;              p += NBCW * 2;   // [B][W][C]
    __bf16* pv    = (__bf16*)p;              p += NBCW * 2;   // [B][C][W] (K=w contiguous)
    __bf16* qeb   = (__bf16*)p;              p += NWW2 * 2;   // [B][i][j]  (K=j contiguous)
    __bf16* keb   = (__bf16*)p;              p += NWW2 * 2;   // [B][j][i]  (K=i contiguous)
    __bf16* attb  = (__bf16*)p;              p += NWW2 * 2;   // [B][j][k]  (K contiguous)
    double* sums  = (double*)p;              p += 4 * 8;
    float*  energy = t0;  // 64MB aliasing t0+xg (both dead by energy time)

    const int T = 256;
    dim3 tpb(32, 8, 1);
    dim3 gTr(WW / 32, CC / 32, BB);

    // 1. inputs: transpose x to [B][W][C] bf16; pack weights
    transpose_cvt_kernel<<<gTr, tpb, 0, stream>>>(x, xT, nullptr);
    pack_conv3_kernel<<<(int)((3LL * CC * CC + T - 1) / T), T, 0, stream>>>(rw1, wr1p);
    pack_conv3_kernel<<<(int)((3LL * CC * CC + T - 1) / T), T, 0, stream>>>(rw2, wr2p);
    cvt_bf16_kernel<<<(int)(((long long)CC * CC + T - 1) / T), T, 0, stream>>>(qw, wq, (long long)CC * CC);
    cvt_bf16_kernel<<<(int)(((long long)CC * CC + T - 1) / T), T, 0, stream>>>(kw, wk, (long long)CC * CC);
    cvt_bf16_kernel<<<(int)(((long long)CC * CC + T - 1) / T), T, 0, stream>>>(vw, wv, (long long)CC * CC);

    // 2. residual block: conv3 -> IN -> sin, twice (conv biases cancel under IN)
    dim3 gConv(WW / 64, CC / 16, BB);
    conv3_wmma_kernel<<<gConv, 32, 0, stream>>>(xT, wr1p, t0);
    in_sin_kernel<<<BB * CC, T, 0, stream>>>(t0, t0);
    transpose_cvt_kernel<<<gTr, tpb, 0, stream>>>(t0, g1T, nullptr);
    conv3_wmma_kernel<<<gConv, 32, 0, stream>>>(g1T, wr2p, t0);
    in_sin_kernel<<<BB * CC, T, 0, stream>>>(t0, g);
    transpose_cvt_kernel<<<gTr, tpb, 0, stream>>>(g, gT, nullptr);

    // 3. x_g = x + g; global z-scores of x_g and g (transposed bf16 outputs)
    add_xg_kernel<<<(int)((NBCW + T - 1) / T), T, 0, stream>>>(x, g, xg, NBCW);
    transpose_cvt_kernel<<<gTr, tpb, 0, stream>>>(xg, xgT, nullptr);
    zero_sums_kernel<<<1, 32, 0, stream>>>(sums);
    reduce_sums_kernel<<<1024, T, 0, stream>>>(xg, NBCW, sums);
    reduce_sums_kernel<<<1024, T, 0, stream>>>(g, NBCW, sums + 2);
    transpose_cvt_kernel<<<gTr, tpb, 0, stream>>>(xg, xzT, sums);
    transpose_cvt_kernel<<<gTr, tpb, 0, stream>>>(g, gzT, sums + 2);

    // 4. 1x1 convs (q/k/v); biases are zero.  D[co,w] = sum_ci Wq[co,ci]*act[w,ci]
    dim3 gProj(WW / 64, CC / 16, BB);
    gemm2_wmma_bf16_kernel<<<gProj, 32, 0, stream>>>(
        wq, 0, CC, xgT, CW, CC, nullptr, pqT, CW, 1, CC, CC);   // store [w][co]
    gemm2_wmma_bf16_kernel<<<gProj, 32, 0, stream>>>(
        wk, 0, CC, gT, CW, CC, nullptr, pkT, CW, 1, CC, CC);    // store [w][co]
    gemm2_wmma_bf16_kernel<<<gProj, 32, 0, stream>>>(
        wv, 0, CC, gT, CW, CC, nullptr, pv, CW, WW, 1, CC);     // store [co][w]

    // 5. qe[b,j,i] = sum_c pqT[j,c]*xzT[i,c]  -> store qe_rm[i][j] (j contiguous)
    dim3 gWW(WW / 64, WW / 16, BB);
    gemm2_wmma_bf16_kernel<<<gWW, 32, 0, stream>>>(
        pqT, CW, CC, xzT, CW, CC, nullptr, qeb, W2, 1, WW, CC);  // D(m=j,n=i) at [i*W+j]
    // 6. ke[b,i,j] = sum_c gzT[i,c]*pkT[j,c]  -> store keT[j][i] (i contiguous)
    gemm2_wmma_bf16_kernel<<<gWW, 32, 0, stream>>>(
        gzT, CW, CC, pkT, CW, CC, nullptr, keb, W2, 1, WW, CC);  // D(m=i,n=j) at [j*W+i]
    // 7. energy[b,i,j] = sum_k qe_rm[i,k]*keT[j,k]
    gemm2_wmma_bf16_kernel<<<gWW, 32, 0, stream>>>(
        qeb, W2, WW, keb, W2, WW, energy, nullptr, W2, WW, 1, WW);
    // 8. softmax rows -> bf16 attention [b][row][col], col contiguous
    softmax_att_kernel<<<BB * WW, T, 0, stream>>>(energy, attb);
    // 9. out[b,c,j] = sum_k pv[c,k]*att[j,k]
    dim3 gOut(WW / 64, CC / 16, BB);
    gemm2_wmma_bf16_kernel<<<gOut, 32, 0, stream>>>(
        pv, CW, WW, attb, W2, WW, out, nullptr, CW, WW, 1, WW);
}
